// GCN_BiLSTM_34368328302738
// MI455X (gfx1250) — compile-verified, matrix operand order checked by reading, not run on hardware
//
#include <hip/hip_runtime.h>
#include <hip/hip_bf16.h>

typedef __attribute__((ext_vector_type(16))) _Float16 v16h;
typedef __attribute__((ext_vector_type(8)))  _Float16 v8h;
typedef __attribute__((ext_vector_type(8)))  float    v8f;

#define HB 128   // LSTM hidden
#define GD 512   // 4*HB gates
#define WS 136   // padded LDS row stride (halves) for w_hh: breaks bank conflicts

// ---------------------------------------------------------------- utilities
__global__ void zero_f32(float* __restrict__ p, long n) {
  long i = (long)blockIdx.x * blockDim.x + threadIdx.x;
  if (i < n) p[i] = 0.f;
}

__global__ void f32_to_f16(const float* __restrict__ in, _Float16* __restrict__ out, long n) {
  long i = (long)blockIdx.x * blockDim.x + threadIdx.x;
  if (i < n) out[i] = (_Float16)in[i];
}

// out[c*rows + r] = in[r*cols + c]  (f32 -> f16 transpose)
__global__ void transpose_to_f16(const float* __restrict__ in, _Float16* __restrict__ out,
                                 int rows, int cols) {
  int i = blockIdx.x * blockDim.x + threadIdx.x;
  if (i >= rows * cols) return;
  int r = i / cols, c = i % cols;
  out[(long)c * rows + r] = (_Float16)in[i];
}

// ------------------------------------------------- layer-0 xg: [T,17]x[512,17]^T
__global__ void xg_input_l0(const float* __restrict__ x, const float* __restrict__ w,
                            const float* __restrict__ b1, const float* __restrict__ b2,
                            float* __restrict__ xg, int T) {
  int g = blockIdx.x * blockDim.x + threadIdx.x;
  int t = g >> 9, j = g & (GD - 1);
  if (t >= T) return;
  float acc = b1[j] + b2[j];
  const float* xr = x + t * 17;
  const float* wr = w + j * 17;
#pragma unroll
  for (int k = 0; k < 17; ++k) acc += xr[k] * wr[k];
  xg[(long)t * GD + j] = acc;
}

// ---------------------------------------- serial LSTM scan, w_hh staged in LDS
__global__ __launch_bounds__(256)
void lstm_scan(const float* __restrict__ xg, const float* __restrict__ whh,
               float* __restrict__ hs, int T, int reverse) {
  __shared__ _Float16 w[GD * WS];            // 139 KB of the 320 KB WGP LDS
  __shared__ float sh[HB], sc[HB], sg[GD];
  int tid = threadIdx.x;
  for (int i = tid; i < GD * HB; i += 256) { // one-time f32->f16 stage
    int row = i >> 7, col = i & (HB - 1);
    w[row * WS + col] = (_Float16)whh[i];
  }
  if (tid < HB) { sh[tid] = 0.f; sc[tid] = 0.f; }
  __syncthreads();
  for (int step = 0; step < T; ++step) {
    int t = reverse ? (T - 1 - step) : step;
    const float* xr = xg + (long)t * GD;
    for (int gi = tid; gi < GD; gi += 256) { // 2 gates / thread
      float acc = xr[gi];
      const v8h* wv = (const v8h*)(w + gi * WS);   // ds_load_b128 chunks
#pragma unroll
      for (int k8 = 0; k8 < 16; ++k8) {
        v8h cw = wv[k8];
#pragma unroll
        for (int j = 0; j < 8; ++j) acc += (float)cw[j] * sh[k8 * 8 + j];
      }
      sg[gi] = acc;
    }
    __syncthreads();
    if (tid < HB) {
      float ig = 1.f / (1.f + __expf(-sg[tid]));
      float fg = 1.f / (1.f + __expf(-sg[HB + tid]));
      float gg = tanhf(sg[2 * HB + tid]);
      float og = 1.f / (1.f + __expf(-sg[3 * HB + tid]));
      float c = fg * sc[tid] + ig * gg;
      sc[tid] = c;
      float h = og * tanhf(c);
      sh[tid] = h;
      hs[(long)t * HB + tid] = h;
    }
    __syncthreads();
  }
}

__global__ void concat_tf_f16(const float* __restrict__ hf, const float* __restrict__ hr,
                              _Float16* __restrict__ x1, int T) {
  int i = blockIdx.x * blockDim.x + threadIdx.x;
  int t = i >> 8, j = i & 255;
  if (t >= T) return;
  x1[(long)t * 256 + j] =
      (_Float16)((j < HB) ? hf[(long)t * HB + j] : hr[(long)t * HB + (j - HB)]);
}

// ---------------------------------------------------------------- WMMA GEMM
// C[M,N] = relu?(A[M,K] * B[N,K]^T + bias1 + bias2); A,B f16 row-major.
// ISA 16-bit fragment layout: lane(0-15) K-base 0, lane(16-31) K-base 8;
// halves 0..7 -> K+0..7, halves 8..15 -> K+16..23  => two b128 loads per frag.
// Each wave computes a 16x32 output (two acc tiles share the A fragment);
// 4 waves/block cover 128 N-columns. M%16==0, N%128==0, K%32==0.
__global__ __launch_bounds__(128)
void wmma_gemm_f16(const _Float16* __restrict__ A, int lda,
                   const _Float16* __restrict__ B, int ldb,
                   const float* __restrict__ bias1, const float* __restrict__ bias2,
                   float* __restrict__ C, int ldc, int K, int do_relu) {
  int lane = threadIdx.x & 31;
  int wave = threadIdx.x >> 5;
  int l15  = lane & 15;
  int hi   = lane >> 4;
  int kb   = hi * 8;
  int n0   = (blockIdx.x * 4 + wave) * 32;
  int m0   = blockIdx.y * 16;
  const _Float16* arow  = A + (long)(m0 + l15) * lda + kb;
  const _Float16* brow0 = B + (long)(n0 + l15) * ldb + kb;
  const _Float16* brow1 = B + (long)(n0 + 16 + l15) * ldb + kb;
  v8f acc0 = {}, acc1 = {};
  for (int k0 = 0; k0 < K; k0 += 32) {
    __builtin_prefetch(arow + k0 + 128, 0, 1);       // global_prefetch_b8
    v8h alo = *(const v8h*)(arow + k0);
    v8h ahi = *(const v8h*)(arow + k0 + 16);
    v8h b0l = *(const v8h*)(brow0 + k0);
    v8h b0h = *(const v8h*)(brow0 + k0 + 16);
    v8h b1l = *(const v8h*)(brow1 + k0);
    v8h b1h = *(const v8h*)(brow1 + k0 + 16);
    v16h af  = __builtin_shufflevector(alo, ahi, 0,1,2,3,4,5,6,7,8,9,10,11,12,13,14,15);
    v16h bf0 = __builtin_shufflevector(b0l, b0h, 0,1,2,3,4,5,6,7,8,9,10,11,12,13,14,15);
    v16h bf1 = __builtin_shufflevector(b1l, b1h, 0,1,2,3,4,5,6,7,8,9,10,11,12,13,14,15);
    acc0 = __builtin_amdgcn_wmma_f32_16x16x32_f16(false, af, false, bf0,
                                                  (short)0, acc0, false, false);
    acc1 = __builtin_amdgcn_wmma_f32_16x16x32_f16(false, af, false, bf1,
                                                  (short)0, acc1, false, false);
  }
  int na = n0 + l15, nb = n0 + 16 + l15;
  float bva = bias1[na] + bias2[na];
  float bvb = bias1[nb] + bias2[nb];
#pragma unroll
  for (int r = 0; r < 8; ++r) {                      // C/D: VGPR r -> M = r + 8*hi
    int m = m0 + r + hi * 8;
    float v0 = acc0[r] + bva;
    float v1 = acc1[r] + bvb;
    if (do_relu) { v0 = v0 > 0.f ? v0 : 0.f; v1 = v1 > 0.f ? v1 : 0.f; }
    C[(long)m * ldc + na] = v0;
    C[(long)m * ldc + nb] = v1;
  }
}

// ------------------------------------------------- time mean of concat(hf,hr)
__global__ void time_mean(const float* __restrict__ hf, const float* __restrict__ hr,
                          float* __restrict__ out, int T) {
  int j = threadIdx.x;                               // 256
  const float* src = (j < HB) ? (hf + j) : (hr + (j - HB));
  float s = 0.f;
  for (int t = 0; t < T; ++t) s += src[(long)t * HB];
  out[j] = s / (float)T;
}

// ------------------------------------------------- weighted edge scatter-add
__global__ void edge_scatter(const int* __restrict__ src, const int* __restrict__ dst,
                             const float* __restrict__ w, const float* __restrict__ feat,
                             float* __restrict__ agg, int nE) {
  long idx = (long)blockIdx.x * blockDim.x + threadIdx.x;
  int e = (int)(idx >> 5);                           // 32 float4 chunks = 128 feats
  if (e >= nE) return;
  int c = (int)(idx & 31);
  int s = src[e], d = dst[e];
  float ww = w[e];
  float4 f = ((const float4*)(feat + (long)s * 128))[c];
  float* ap = agg + (long)d * 128 + c * 4;
  unsafeAtomicAdd(ap + 0, ww * f.x);
  unsafeAtomicAdd(ap + 1, ww * f.y);
  unsafeAtomicAdd(ap + 2, ww * f.z);
  unsafeAtomicAdd(ap + 3, ww * f.w);
}

// ------------------------------------------------- column sums of [rows,128]
__global__ void col_sum(const float* __restrict__ a, float* __restrict__ out, int rows) {
  int col = threadIdx.x;                             // 128
  float s = 0.f;
  for (int r = blockIdx.x; r < rows; r += gridDim.x) s += a[(long)r * 128 + col];
  unsafeAtomicAdd(out + col, s);
}

// hg = (colsum/N) @ W2[128,256] + b2
__global__ void hg_linear(const float* __restrict__ cs, const float* __restrict__ W2,
                          const float* __restrict__ b2, float* __restrict__ hg, float invN) {
  int j = threadIdx.x;                               // 256
  float acc = b2[j];
  for (int k = 0; k < 128; ++k) acc += cs[k] * invN * W2[k * 256 + j];
  hg[j] = acc;
}

__global__ void concat_features(const float* __restrict__ hg, const float* __restrict__ gv,
                                const float* __restrict__ kmer, float* __restrict__ f) {
  int i = blockIdx.x * blockDim.x + threadIdx.x;
  if (i < 256)       f[i] = hg[i];
  else if (i < 512)  f[i] = gv[i - 256];
  else if (i < 1150) f[i] = kmer[i - 512];
}

__global__ void mlp_layer(const float* __restrict__ in, const float* __restrict__ W,
                          const float* __restrict__ b, float* __restrict__ out,
                          int O, int I, int do_relu) {
  int o = blockIdx.x * blockDim.x + threadIdx.x;
  if (o >= O) return;
  float acc = b[o];
  const float* wr = W + (long)o * I;
  for (int k = 0; k < I; ++k) acc += wr[k] * in[k];
  if (do_relu && acc < 0.f) acc = 0.f;
  out[o] = acc;
}

extern "C" void kernel_launch(void* const* d_in, const int* in_sizes, int n_in,
                              void* d_out, int out_size, void* d_ws, size_t ws_size,
                              hipStream_t stream) {
  const float* g_mol = (const float*)d_in[0];
  const float* feat  = (const float*)d_in[1];
  const int*   esrc  = (const int*)d_in[2];
  const int*   edst  = (const int*)d_in[3];
  const float* ew    = (const float*)d_in[4];
  const float* kmer  = (const float*)d_in[5];
  const float *w_ih_l0  = (const float*)d_in[6],  *w_hh_l0  = (const float*)d_in[7];
  const float *b_ih_l0  = (const float*)d_in[8],  *b_hh_l0  = (const float*)d_in[9];
  const float *w_ih_l0r = (const float*)d_in[10], *w_hh_l0r = (const float*)d_in[11];
  const float *b_ih_l0r = (const float*)d_in[12], *b_hh_l0r = (const float*)d_in[13];
  const float *w_ih_l1  = (const float*)d_in[14], *w_hh_l1  = (const float*)d_in[15];
  const float *b_ih_l1  = (const float*)d_in[16], *b_hh_l1  = (const float*)d_in[17];
  const float *w_ih_l1r = (const float*)d_in[18], *w_hh_l1r = (const float*)d_in[19];
  const float *b_ih_l1r = (const float*)d_in[20], *b_hh_l1r = (const float*)d_in[21];
  const float *W1 = (const float*)d_in[22], *b1 = (const float*)d_in[23];
  const float *W2 = (const float*)d_in[24], *b2 = (const float*)d_in[25];
  const float *fc1w = (const float*)d_in[26], *fc1b = (const float*)d_in[27];
  const float *fc2w = (const float*)d_in[28], *fc2b = (const float*)d_in[29];
  const float *fc3w = (const float*)d_in[30], *fc3b = (const float*)d_in[31];
  const float *fc4w = (const float*)d_in[32], *fc4b = (const float*)d_in[33];

  const int T = in_sizes[0] / 17;
  const int N = in_sizes[1] / 128;
  const int E = in_sizes[2];

  // ---- workspace carve-up: f32 region, then f16 staging region ----
  float* p = (float*)d_ws;
  float* xgf   = p; p += (long)T * GD;       // reused: layer0 then layer1
  float* xgr   = p; p += (long)T * GD;
  float* hs0f  = p; p += (long)T * HB;
  float* hs0r  = p; p += (long)T * HB;
  float* hs1f  = p; p += (long)T * HB;
  float* hs1r  = p; p += (long)T * HB;
  float* gvec  = p; p += 256;
  float* agg   = p; p += (long)N * 128;      // reused: agg1 then agg2
  float* hbuf  = p; p += (long)N * 128;
  float* csum  = p; p += 128;
  float* hg    = p; p += 256;
  float* fts   = p; p += 1152;
  float* m1    = p; p += 576;
  float* m2    = p; p += 256;
  float* m3    = p; p += 64;
  float* zerov = p; p += 512;
  _Float16* q = (_Float16*)p;                // 16B-aligned (all counts %4==0)
  _Float16* x1h    = q; q += (long)T * 256;
  _Float16* wih1h  = q; q += GD * 256;
  _Float16* wih1rh = q; q += GD * 256;
  _Float16* aggh   = q; q += (long)N * 128;
  _Float16* w1th   = q; q += 128 * 128;

  // ---- one-time-per-call f16 weight staging ----
  zero_f32<<<2, 256, 0, stream>>>(zerov, 512);
  f32_to_f16<<<(GD * 256 + 255) / 256, 256, 0, stream>>>(w_ih_l1,  wih1h,  GD * 256);
  f32_to_f16<<<(GD * 256 + 255) / 256, 256, 0, stream>>>(w_ih_l1r, wih1rh, GD * 256);
  transpose_to_f16<<<(128 * 128 + 255) / 256, 256, 0, stream>>>(W1, w1th, 128, 128);

  // ---------------- BiLSTM layer 0 ----------------
  int xg_blocks = (T * GD + 255) / 256;
  xg_input_l0<<<xg_blocks, 256, 0, stream>>>(g_mol, w_ih_l0,  b_ih_l0,  b_hh_l0,  xgf, T);
  xg_input_l0<<<xg_blocks, 256, 0, stream>>>(g_mol, w_ih_l0r, b_ih_l0r, b_hh_l0r, xgr, T);
  lstm_scan<<<1, 256, 0, stream>>>(xgf, w_hh_l0,  hs0f, T, 0);
  lstm_scan<<<1, 256, 0, stream>>>(xgr, w_hh_l0r, hs0r, T, 1);
  concat_tf_f16<<<(T * 256 + 255) / 256, 256, 0, stream>>>(hs0f, hs0r, x1h, T);

  // ---------------- BiLSTM layer 1 (xg via WMMA: x1h[T,256] @ w_ih[512,256]^T) --
  dim3 g1(GD / 128, T / 16);
  wmma_gemm_f16<<<g1, 128, 0, stream>>>(x1h, 256, wih1h,  256, b_ih_l1,  b_hh_l1,  xgf, GD, 256, 0);
  wmma_gemm_f16<<<g1, 128, 0, stream>>>(x1h, 256, wih1rh, 256, b_ih_l1r, b_hh_l1r, xgr, GD, 256, 0);
  lstm_scan<<<1, 256, 0, stream>>>(xgf, w_hh_l1,  hs1f, T, 0);
  lstm_scan<<<1, 256, 0, stream>>>(xgr, w_hh_l1r, hs1r, T, 1);
  time_mean<<<1, 256, 0, stream>>>(hs1f, hs1r, gvec, T);

  // ---------------- GraphConv 1 ----------------
  long aggN = (long)N * 128;
  zero_f32<<<(int)((aggN + 255) / 256), 256, 0, stream>>>(agg, aggN);
  edge_scatter<<<(int)(((long)E * 32 + 255) / 256), 256, 0, stream>>>(esrc, edst, ew, feat, agg, E);
  f32_to_f16<<<(int)((aggN + 255) / 256), 256, 0, stream>>>(agg, aggh, aggN);
  dim3 gc1(1, N / 16);   // N columns = 128
  wmma_gemm_f16<<<gc1, 128, 0, stream>>>(aggh, 128, w1th, 128, b1, zerov, hbuf, 128, 128, 1);

  // ---------------- GraphConv 2 (mean folded through linearity) ----------------
  zero_f32<<<(int)((aggN + 255) / 256), 256, 0, stream>>>(agg, aggN);
  edge_scatter<<<(int)(((long)E * 32 + 255) / 256), 256, 0, stream>>>(esrc, edst, ew, hbuf, agg, E);
  zero_f32<<<1, 128, 0, stream>>>(csum, 128);
  col_sum<<<64, 128, 0, stream>>>(agg, csum, N);
  hg_linear<<<1, 256, 0, stream>>>(csum, W2, b2, hg, 1.0f / (float)N);

  // ---------------- head ----------------
  concat_features<<<5, 256, 0, stream>>>(hg, gvec, kmer, fts);
  mlp_layer<<<3, 256, 0, stream>>>(fts, fc1w, fc1b, m1, 575, 1150, 1);
  mlp_layer<<<1, 256, 0, stream>>>(m1, fc2w, fc2b, m2, 256, 575, 1);
  mlp_layer<<<1, 64, 0, stream>>>(m2, fc3w, fc3b, m3, 64, 256, 1);
  mlp_layer<<<1, 64, 0, stream>>>(m3, fc4w, fc4b, (float*)d_out, 1, 64, 0);
}
